// MlpNet_39848706573752
// MI455X (gfx1250) — compile-verified, matrix operand order checked by reading, not run on hardware
//
#include <hip/hip_runtime.h>
#include <math.h>

// ---------------- Problem constants ----------------
#define BATCH   16
#define NSIG    1023          // signal length
#define TF      1024          // frames (centers 0..1023)
#define FB      1024          // rfft(2046) bins kept
#define KPAD    1024          // K dim (n=0..1022, pad 1023 with zero)
#define MROWS   (BATCH * TF)  // 16384 GEMM rows
#define HID     128
#define NCLS    10
#define FEAT    (FB * TF)     // 1048576

typedef __attribute__((ext_vector_type(16))) __bf16 v16bf;
typedef __attribute__((ext_vector_type(8)))  float  v8f;
typedef __attribute__((ext_vector_type(4)))  unsigned u32x4;
typedef __attribute__((ext_vector_type(8)))  int      i32x8;
typedef __attribute__((ext_vector_type(4)))  int      i32x4;

union FragU {
    v16bf v;
    uint4 q[2];
};

// One pipeline stage: A frag + 2 f-tiles of (cos,sin) B frags
struct Stage {
    FragU a;
    FragU bc0, bs0, bc1, bs1;
};

// round-to-nearest-even float -> bf16 bits
static __device__ __forceinline__ unsigned short f2bf(float f) {
    union { float f; unsigned u; } x;
    x.f = f;
    unsigned r = x.u + 0x7FFFu + ((x.u >> 16) & 1u);
    return (unsigned short)(r >> 16);
}

// ---------------- 1. per-sample mean removal -> xc[16][1024] (f32, n=1023 padded 0)
__global__ void mean_xc_kernel(const float* __restrict__ x, float* __restrict__ xc) {
    __shared__ float red[256];
    const int b = blockIdx.x;
    const int tid = threadIdx.x;
    float s = 0.f;
    for (int n = tid; n < NSIG; n += 256) s += x[b * NSIG + n];
    red[tid] = s;
    __syncthreads();
    for (int off = 128; off > 0; off >>= 1) {
        if (tid < off) red[tid] += red[tid + off];
        __syncthreads();
    }
    const float mean = red[0] * (1.0f / (float)NSIG);
    for (int n = tid; n < KPAD; n += 256)
        xc[b * KPAD + n] = (n < NSIG) ? (x[b * NSIG + n] - mean) : 0.0f;
}

// ---------------- 2. A[row=b*T+t][n] = bf16( xc[b][n] * exp(-0.5*((n-t)/sigma)^2) )
__global__ void build_a_kernel(const float* __restrict__ xc,
                               const float* __restrict__ lambd,
                               unsigned short* __restrict__ A) {
    const int row = blockIdx.x;          // 0..16383
    const int b = row >> 10;
    const int t = row & (TF - 1);
    const float sigma = fabsf(lambd[0]);
    const float inv_s = 1.0f / sigma;
    for (int i = 0; i < 4; ++i) {
        const int n = i * 256 + threadIdx.x;
        const float d = ((float)n - (float)t) * inv_s;
        const float w = __expf(-0.5f * d * d);
        A[(size_t)row * KPAD + n] = f2bf(xc[b * KPAD + n] * w);
    }
}

// ---------------- 3. DFT basis (symmetric in k<->f): Bc[k][f]=cos, Bs[k][f]=-sin
__global__ void build_b_kernel(unsigned short* __restrict__ Bc,
                               unsigned short* __restrict__ Bs) {
    const int k = blockIdx.x;            // 0..1023
    const float w0 = (float)(2.0 * M_PI / (double)(2 * NSIG));  // 2pi/2046
    for (int i = 0; i < 4; ++i) {
        const int f = i * 256 + threadIdx.x;
        const float theta = w0 * (float)k * (float)f;
        float sv, cv;
        __sincosf(theta, &sv, &cv);
        Bc[(size_t)k * FB + f] = f2bf(cv);
        Bs[(size_t)k * FB + f] = f2bf(-sv);
    }
}

// ---------------- 4. Spectrogram GEMM via bf16 WMMA, fused |.|^2 store
// All 8 waves of a block share one 16-row M-tile staged in LDS by the TDM.
// Each wave: 2 f-tiles, software-pipelined (double-buffered Stage structs) so
// next-k loads are in flight while current WMMAs occupy the XDL pipe.
__global__ void spect_wmma_kernel(const unsigned short* __restrict__ A,
                                  const unsigned short* __restrict__ Bc,
                                  const unsigned short* __restrict__ Bs,
                                  float* __restrict__ sOut) {
    __shared__ unsigned short aLds[16 * KPAD];   // 32 KB A tile

    const int wid  = threadIdx.x >> 5;
    const int lane = threadIdx.x & 31;
    const int wg   = blockIdx.x * 8 + wid;   // 0..32767 waves
    const int mtile = wg >> 5;               // constant across the block
    const int fgrp  = wg & 31;               // 0..31 -> 32 f's per wave
    const int half  = lane >> 4;
    const int lrow  = lane & 15;

    // ---- TDM: DMA A tile [16 x 1024] bf16 from global into LDS (wave 0 only)
    if (wid == 0) {
        const unsigned long long gaddr =
            (unsigned long long)(const void*)(A + (size_t)mtile * 16 * KPAD);
        const unsigned ldsaddr = (unsigned)(unsigned long long)(const void*)aLds;
        u32x4 g0;
        g0[0] = 1u;                                   // count=1, is_restore=0
        g0[1] = ldsaddr;                              // lds_addr [63:32]
        g0[2] = (unsigned)(gaddr & 0xFFFFFFFFu);      // global_addr lo
        g0[3] = (unsigned)((gaddr >> 32) & 0x01FFFFFFu) | (2u << 30);  // hi | type=2
        i32x8 g1;
        g1[0] = (int)(1u << 16);                      // wg_mask=0, data_size=1 (2B)
        g1[1] = (int)((KPAD & 0xFFFFu) << 16);        // tensor_dim0[15:0]=1024
        g1[2] = (int)(16u << 16);                     // tensor_dim0 hi=0 | tensor_dim1=16
        g1[3] = (int)((unsigned)KPAD << 16);          // tensor_dim1 hi=0 | tile_dim0=1024
        g1[4] = 16;                                   // tile_dim1=16, tile_dim2=0
        g1[5] = KPAD;                                 // tensor_dim0_stride lo = 1024
        g1[6] = 0;                                    // stride hi | tensor_dim1_stride lo
        g1[7] = 0;
        i32x4 gz4 = {0, 0, 0, 0};
        i32x8 gz8 = {0, 0, 0, 0, 0, 0, 0, 0};
        __builtin_amdgcn_tensor_load_to_lds(g0, g1, gz4, gz4, gz8, 0);
        __builtin_amdgcn_s_wait_tensorcnt(0);
    }
    __syncthreads();

    const unsigned short* arow = aLds + lrow * KPAD;
    const int f0 = fgrp * 32 + lrow;                 // f-tile 0 basis row
    const int f1 = f0 + 16;                          // f-tile 1 basis row
    const unsigned short* bcr0 = Bc + (size_t)f0 * KPAD;
    const unsigned short* bsr0 = Bs + (size_t)f0 * KPAD;
    const unsigned short* bcr1 = Bc + (size_t)f1 * KPAD;
    const unsigned short* bsr1 = Bs + (size_t)f1 * KPAD;
    const int o0 = half * 8;                         // frag chunk offsets (elems)
    const int o1 = 16 + half * 8;

    v8f accRe0 = v8f{}, accIm0 = v8f{}, accRe1 = v8f{}, accIm1 = v8f{};

    Stage F0, F1;

#define LOAD_STAGE(F, kk)                                              \
    do {                                                               \
        (F).a.q[0]   = *(const uint4*)(arow + (kk) + o0);              \
        (F).a.q[1]   = *(const uint4*)(arow + (kk) + o1);              \
        (F).bc0.q[0] = *(const uint4*)(bcr0 + (kk) + o0);              \
        (F).bc0.q[1] = *(const uint4*)(bcr0 + (kk) + o1);              \
        (F).bs0.q[0] = *(const uint4*)(bsr0 + (kk) + o0);              \
        (F).bs0.q[1] = *(const uint4*)(bsr0 + (kk) + o1);              \
        (F).bc1.q[0] = *(const uint4*)(bcr1 + (kk) + o0);              \
        (F).bc1.q[1] = *(const uint4*)(bcr1 + (kk) + o1);              \
        (F).bs1.q[0] = *(const uint4*)(bsr1 + (kk) + o0);              \
        (F).bs1.q[1] = *(const uint4*)(bsr1 + (kk) + o1);              \
    } while (0)

#define COMPUTE_STAGE(F)                                               \
    do {                                                               \
        accRe0 = __builtin_amdgcn_wmma_f32_16x16x32_bf16(              \
            false, (F).a.v, false, (F).bc0.v, (short)0, accRe0, false, false); \
        accIm0 = __builtin_amdgcn_wmma_f32_16x16x32_bf16(              \
            false, (F).a.v, false, (F).bs0.v, (short)0, accIm0, false, false); \
        accRe1 = __builtin_amdgcn_wmma_f32_16x16x32_bf16(              \
            false, (F).a.v, false, (F).bc1.v, (short)0, accRe1, false, false); \
        accIm1 = __builtin_amdgcn_wmma_f32_16x16x32_bf16(              \
            false, (F).a.v, false, (F).bs1.v, (short)0, accIm1, false, false); \
    } while (0)

    LOAD_STAGE(F0, 0);
#pragma unroll 1
    for (int kk = 0; kk < KPAD - 64; kk += 64) {
        LOAD_STAGE(F1, kk + 32);
        COMPUTE_STAGE(F0);
        LOAD_STAGE(F0, kk + 64);
        COMPUTE_STAGE(F1);
    }
    LOAD_STAGE(F1, KPAD - 32);
    COMPUTE_STAGE(F0);
    COMPUTE_STAGE(F1);

#undef LOAD_STAGE
#undef COMPUTE_STAGE

    // Store S = Re^2 + Im^2 ; C/D layout: M = r + 8*(lane>=16), N = lane%16
#pragma unroll
    for (int r = 0; r < 8; ++r) {
        const int row = mtile * 16 + r + half * 8;
        const int b = row >> 10;
        const int t = row & (TF - 1);
        const float re0 = accRe0[r], im0 = accIm0[r];
        const float re1 = accRe1[r], im1 = accIm1[r];
        sOut[(size_t)b * FEAT + (size_t)f0 * TF + t] = re0 * re0 + im0 * im0;
        sOut[(size_t)b * FEAT + (size_t)f1 * TF + t] = re1 * re1 + im1 * im1;
    }
}

// ---------------- 5. zero hacc
__global__ void zero_hacc_kernel(float* __restrict__ hacc) {
    const int i = blockIdx.x * 256 + threadIdx.x;
    if (i < BATCH * HID) hacc[i] = 0.0f;
}

// ---------------- 6. fc1 split-K: hacc[b][hid] += sum_k W1[hid,k]*s[b,k]
// HBM-bound on W1 (512 MB); coalesced streaming + LDS-staged s-slice + shuffle reduce.
__global__ void fc1_splitk_kernel(const float* __restrict__ s,
                                  const float* __restrict__ W1,
                                  float* __restrict__ hacc) {
    __shared__ float slds[BATCH * 1024];     // 64 KB slice of s
    const int tid = threadIdx.x;
    const size_t kbase = (size_t)blockIdx.x * 1024;

    // stage s[:, kbase:kbase+1024] (4096 float4, 16 per thread, coalesced)
    const float4* s4 = (const float4*)s;
    float4* l4 = (float4*)slds;
#pragma unroll
    for (int i = 0; i < 16; ++i) {
        const int j = tid + i * 256;         // 0..4095
        const int b = j >> 8;                // 256 float4 per row
        const int kq = j & 255;
        l4[b * 256 + kq] = s4[(size_t)b * (FEAT / 4) + (kbase >> 2) + kq];
    }
    __syncthreads();

    const int wid = tid >> 5;
    const int lane = tid & 31;
#pragma unroll 1
    for (int hh = 0; hh < 16; ++hh) {
        const int hid = wid * 16 + hh;
        const float* wrow = W1 + (size_t)hid * FEAT + kbase;
        __builtin_prefetch(wrow + lane * 4, 0, 1);   // global_prefetch_b8
        float acc[BATCH];
#pragma unroll
        for (int b = 0; b < BATCH; ++b) acc[b] = 0.0f;
#pragma unroll
        for (int i = 0; i < 32; ++i) {
            const int k = i * 32 + lane;
            const float w = wrow[k];
#pragma unroll
            for (int b = 0; b < BATCH; ++b) acc[b] += w * slds[b * 1024 + k];
        }
#pragma unroll
        for (int off = 16; off > 0; off >>= 1)
#pragma unroll
            for (int b = 0; b < BATCH; ++b)
                acc[b] += __shfl_xor(acc[b], off, 32);
        if (lane == 0) {
#pragma unroll
            for (int b = 0; b < BATCH; ++b)
                atomicAdd(&hacc[b * HID + hid], acc[b]);
        }
    }
}

// ---------------- 7. fc2: out = relu(hacc+b1) @ W2^T + b2
__global__ void fc2_kernel(const float* __restrict__ hacc,
                           const float* __restrict__ b1,
                           const float* __restrict__ W2,
                           const float* __restrict__ b2,
                           float* __restrict__ out) {
    __shared__ float h[BATCH * HID];
    const int tid = threadIdx.x;
    for (int i = tid; i < BATCH * HID; i += 256) {
        const float v = hacc[i] + b1[i & (HID - 1)];
        h[i] = v > 0.0f ? v : 0.0f;
    }
    __syncthreads();
    if (tid < BATCH * NCLS) {
        const int b = tid / NCLS;
        const int c = tid % NCLS;
        float acc = b2[c];
#pragma unroll 4
        for (int k = 0; k < HID; ++k) acc += h[b * HID + k] * W2[c * HID + k];
        out[b * NCLS + c] = acc;
    }
}

extern "C" void kernel_launch(void* const* d_in, const int* in_sizes, int n_in,
                              void* d_out, int out_size, void* d_ws, size_t ws_size,
                              hipStream_t stream) {
    const float* x     = (const float*)d_in[0];
    const float* lambd = (const float*)d_in[1];
    const float* W1    = (const float*)d_in[2];
    const float* b1    = (const float*)d_in[3];
    const float* W2    = (const float*)d_in[4];
    const float* b2    = (const float*)d_in[5];

    float* out  = (float*)d_out;                  // [16,10]
    float* sOut = (float*)d_out + BATCH * NCLS;   // [16,1,1024,1024]

    char* ws = (char*)d_ws;
    float*          xc   = (float*)ws;                                  // 64 KB
    unsigned short* A    = (unsigned short*)(ws + (1u << 16));          // 32 MB
    unsigned short* Bc   = (unsigned short*)(ws + (1u << 16) + (32u << 20));
    unsigned short* Bs   = (unsigned short*)(ws + (1u << 16) + (32u << 20) + (2u << 20));
    float*          hacc = (float*)(ws + (1u << 16) + (32u << 20) + (4u << 20));

    mean_xc_kernel   <<<BATCH, 256, 0, stream>>>(x, xc);
    build_a_kernel   <<<MROWS, 256, 0, stream>>>(xc, lambd, A);
    build_b_kernel   <<<KPAD,  256, 0, stream>>>(Bc, Bs);
    spect_wmma_kernel<<<MROWS / 4 /*4096 blocks, 8 waves*/, 256, 0, stream>>>(A, Bc, Bs, sOut);
    zero_hacc_kernel <<<(BATCH * HID + 255) / 256, 256, 0, stream>>>(hacc);
    fc1_splitk_kernel<<<FEAT / 1024 /*1024 chunks*/, 256, 0, stream>>>(sOut, W1, hacc);
    fc2_kernel       <<<1, 256, 0, stream>>>(hacc, b1, W2, b2, out);
}
